// MoELayer_58222576665017
// MI455X (gfx1250) — matrix operand order repaired and checked by reading, not compiled, not run on hardware
//
#include <hip/hip_runtime.h>
#include <hip/hip_bf16.h>
#include <math.h>

// ---------------------------------------------------------------------------
// MoE layer (shared + top-2 routed SwiGLU experts) for gfx1250.
// bf16 WMMA grouped GEMM; TDM (tensor_load_to_lds) stages contiguous A tiles.
// ---------------------------------------------------------------------------

typedef __bf16 bf16;
typedef __attribute__((ext_vector_type(16))) __bf16 bf16x16;
typedef __attribute__((ext_vector_type(8)))  __bf16 bf16x8;
typedef __attribute__((ext_vector_type(4)))  __bf16 bf16x4;
typedef __attribute__((ext_vector_type(8)))  float  f32x8;

#define T_TOKENS 8192
#define DMODEL   1024
#define DFF      2048
#define NEXP     16

#define BM  128   // block tile rows (tokens)
#define BN  64    // block tile cols
#define KB  32    // K step (bf16 WMMA K)
#define KBP 40    // padded LDS K-stride (80B: odd bank multiplier, 16B aligned)

#if defined(__gfx1250__) && __has_builtin(__builtin_amdgcn_tensor_load_to_lds)
#define HAVE_TDM 1
#else
#define HAVE_TDM 0
#endif

#if HAVE_TDM
typedef __attribute__((ext_vector_type(4))) unsigned int u32x4;
typedef __attribute__((ext_vector_type(4))) int          i32x4;
typedef __attribute__((ext_vector_type(8))) int          i32x8;

// DMA one [vrows x KB] bf16 tile (row stride `row_stride` elements) into LDS
// at byte offset lds_off, padding each 64B row by 16B -> KBP=40 element rows.
// OOB rows (y >= vrows) are zero-filled by the TDM.
__device__ __forceinline__ void tdm_load_a(const bf16* gsrc, unsigned lds_off,
                                           unsigned row_stride, unsigned vrows) {
  const unsigned long long ga = (unsigned long long)(uintptr_t)gsrc;
  u32x4 g0;
  g0[0] = 1u;                                    // count=1 user descriptor
  g0[1] = lds_off;                               // lds_addr (bytes)
  g0[2] = (unsigned)(ga & 0xFFFFFFFFull);        // global_addr[31:0]
  g0[3] = (unsigned)((ga >> 32) & 0x1FFFFFFu)    // global_addr[56:32]
        | (2u << 30);                            // type = 2 ("image")
  i32x8 g1;
  g1[0] = (1 << 16)                              // data_size = 2 bytes
        | (1 << 20)                              // pad_enable
        | (3 << 22)                              // pad_interval: 16 DWORDs
        | (3 << 25);                             // pad_amount:   4 DWORDs
  g1[1] = (int)((unsigned)KB << 16);             // tensor_dim0 = KB (lo16)
  g1[2] = (int)((vrows & 0xFFFFu) << 16);        // dim0 hi16=0 | tensor_dim1 lo16
  g1[3] = (int)(((vrows >> 16) & 0xFFFFu)        // tensor_dim1 hi16
        | ((unsigned)KB << 16));                 // tile_dim0 = KB
  g1[4] = BM;                                    // tile_dim1 = BM, tile_dim2 = 0
  g1[5] = (int)row_stride;                       // tensor_dim0_stride lo32
  g1[6] = 0;                                     // stride hi16 | dim1_stride lo16
  g1[7] = 0;
  const i32x4 z4 = {0, 0, 0, 0};
#if __clang_major__ >= 23
  const i32x8 z8 = {0, 0, 0, 0, 0, 0, 0, 0};
  __builtin_amdgcn_tensor_load_to_lds(g0, g1, z4, z4, z8, 0);
#else
  __builtin_amdgcn_tensor_load_to_lds(g0, g1, z4, z4, 0);
#endif
}
#endif  // HAVE_TDM

// ---- fragment loads matching CDNA5 16-bit WMMA VGPR layouts -----------------
__device__ __forceinline__ bf16x16 ldsA_frag(const bf16* s, int row0) {
  const int lane = threadIdx.x & 31;
  const int m    = lane & 15;
  const int kh   = (lane >> 4) << 3;
  const bf16* p  = s + (row0 + m) * KBP;
  bf16x8 lo = *(const bf16x8*)(p + kh);
  bf16x8 hi = *(const bf16x8*)(p + 16 + kh);
  bf16x16 f;
#pragma unroll
  for (int i = 0; i < 8; ++i) { f[i] = lo[i]; f[i + 8] = hi[i]; }
  return f;
}

__device__ __forceinline__ bf16x16 ldsB_frag(const bf16* s, int col0) {
  const int lane = threadIdx.x & 31;
  const int n    = lane & 15;
  const int kq   = (lane >> 4) << 3;
  const bf16* p  = s + (col0 + n) * KBP + kq;
  bf16x8 lo = *(const bf16x8*)(p);
  bf16x8 hi = *(const bf16x8*)(p + 16);
  bf16x16 f;
#pragma unroll
  for (int i = 0; i < 8; ++i) { f[i] = lo[i]; f[i + 8] = hi[i]; }
  return f;
}

__device__ __forceinline__ f32x8 wmma_bf16(bf16x16 a, bf16x16 b, f32x8 c) {
  return __builtin_amdgcn_wmma_f32_16x16x32_bf16(false, a, false, b,
                                                 (short)0, c, false, false);
}

// stage a [KB x BN] fp32 weight tile as bf16, K-contiguous per column, using
// 4x float2 coalesced loads and ds_store_b64-wide packed stores.
__device__ __forceinline__ void stageB(const float* __restrict__ w, int ldw,
                                       bf16* __restrict__ sB, int tid) {
  const int n2 = (tid & 31) * 2;   // 2 columns
  const int kq = (tid >> 5) * 4;   // 4 k-rows
  const float* src = w + (size_t)kq * ldw + n2;
  const float2 r0 = *(const float2*)(src);
  const float2 r1 = *(const float2*)(src + ldw);
  const float2 r2 = *(const float2*)(src + 2 * (size_t)ldw);
  const float2 r3 = *(const float2*)(src + 3 * (size_t)ldw);
  bf16x4 c0, c1;
  c0[0] = (bf16)r0.x; c0[1] = (bf16)r1.x; c0[2] = (bf16)r2.x; c0[3] = (bf16)r3.x;
  c1[0] = (bf16)r0.y; c1[1] = (bf16)r1.y; c1[2] = (bf16)r2.y; c1[3] = (bf16)r3.y;
  *(bf16x4*)(sB + (n2 + 0) * KBP + kq) = c0;
  *(bf16x4*)(sB + (n2 + 1) * KBP + kq) = c1;
}

// ---------------------------------------------------------------------------
// small utility kernels
// ---------------------------------------------------------------------------
__global__ void zero_counts_kernel(int* counts) {
  if (threadIdx.x < NEXP) counts[threadIdx.x] = 0;
}

__global__ void cvt_x_kernel(const float* __restrict__ x, bf16* __restrict__ xb,
                             int n4) {
  int i = blockIdx.x * blockDim.x + threadIdx.x;
  if (i < n4) {
    const float4 v = *(const float4*)(x + (size_t)i * 4);
    bf16x4 o;
    o[0] = (bf16)v.x; o[1] = (bf16)v.y; o[2] = (bf16)v.z; o[3] = (bf16)v.w;
    *(bf16x4*)(xb + (size_t)i * 4) = o;
  }
}

__global__ void scan_kernel(const int* __restrict__ counts, int* __restrict__ offs) {
  if (threadIdx.x == 0 && blockIdx.x == 0) {
    int run = 0;
    for (int e = 0; e < NEXP; ++e) { offs[e] = run; run += counts[e]; }
  }
}

__global__ void combine_kernel(float* __restrict__ out,
                               const float* __restrict__ r0,
                               const float* __restrict__ r1, int n) {
  int i = blockIdx.x * blockDim.x + threadIdx.x;
  if (i < n) out[i] = out[i] + r0[i] + r1[i];
}

// ---------------------------------------------------------------------------
// router: one wave32 per token; renormalized top-2 softmax == 2-way softmax
// over the top-2 logits. list entry = token*2 + rank.
// ---------------------------------------------------------------------------
__global__ __launch_bounds__(256)
void router_kernel(const float* __restrict__ x, const float* __restrict__ Wr,
                   float* __restrict__ wts, int* __restrict__ list,
                   int* __restrict__ counts) {
  const int t = blockIdx.x * 8 + (threadIdx.x >> 5);
  if (t >= T_TOKENS) return;
  const int lane = threadIdx.x & 31;

  float acc[NEXP];
#pragma unroll
  for (int e = 0; e < NEXP; ++e) acc[e] = 0.f;

  const float* xr = x + (size_t)t * DMODEL;
  for (int k = lane; k < DMODEL; k += 32) {
    const float xv = xr[k];
    const float* wr = Wr + (size_t)k * NEXP;
#pragma unroll
    for (int e = 0; e < NEXP; ++e) acc[e] += xv * wr[e];
  }
#pragma unroll
  for (int e = 0; e < NEXP; ++e) {
    float v = acc[e];
#pragma unroll
    for (int off = 16; off > 0; off >>= 1) v += __shfl_xor(v, off, 32);
    acc[e] = v;
  }
  if (lane == 0) {
    int   i1 = 0, i2 = -1;
    float l1 = acc[0], l2 = -3.402823466e38f;
#pragma unroll
    for (int e = 1; e < NEXP; ++e) {
      float v = acc[e];
      if (v > l1)      { l2 = l1; i2 = i1; l1 = v; i1 = e; }
      else if (v > l2) { l2 = v; i2 = e; }
    }
    const float w1 = 1.f / (1.f + __expf(l2 - l1));
    wts[t * 2]     = w1;
    wts[t * 2 + 1] = 1.f - w1;
    int s1 = atomicAdd(&counts[i1], 1);
    list[i1 * T_TOKENS + s1] = t * 2;
    int s2 = atomicAdd(&counts[i2], 1);
    list[i2 * T_TOKENS + s2] = t * 2 + 1;
  }
}

// ---------------------------------------------------------------------------
// fused gate+up GEMM: h = silu(A@Wg) * (A@Wu). Shared variant stages A tiles
// with the TDM; routed variant gathers rows through `list`.
// ---------------------------------------------------------------------------
template <bool ROUTED>
__global__ __launch_bounds__(256)
void gateup_kernel(const bf16* __restrict__ xb,
                   const float* __restrict__ Wg, const float* __restrict__ Wu,
                   bf16* __restrict__ hb,
                   const int* __restrict__ list, const int* __restrict__ offs,
                   const int* __restrict__ counts) {
  const int e    = ROUTED ? blockIdx.z : 0;
  const int rows = ROUTED ? counts[e] : T_TOKENS;
  const int row0 = blockIdx.x * BM;
  if (row0 >= rows) return;
  const int n0 = blockIdx.y * BN;

  const float* wg = Wg + (ROUTED ? (size_t)e * DMODEL * DFF : 0);
  const float* wu = Wu + (ROUTED ? (size_t)e * DMODEL * DFF : 0);
  bf16* hbase     = hb + (ROUTED ? (size_t)offs[e] * DFF : 0);
  const int* lst  = ROUTED ? (list + e * T_TOKENS) : nullptr;

  __shared__ bf16 sA[BM * KBP];
  __shared__ bf16 sBg[BN * KBP];
  __shared__ bf16 sBu[BN * KBP];
  __shared__ int  srow[BM];

  const int tid = threadIdx.x;
  if (tid < BM) {
    const int r = row0 + tid;
    srow[tid] = (r < rows) ? (ROUTED ? (lst[r] >> 1) : r) : -1;
  }

  const int wave = tid >> 5;
  const int wm = (wave >> 1) * 32;
  const int wn = (wave & 1) * 32;

  f32x8 accG[2][2], accU[2][2];
#pragma unroll
  for (int i = 0; i < 2; ++i)
#pragma unroll
    for (int j = 0; j < 2; ++j)
#pragma unroll
      for (int q = 0; q < 8; ++q) { accG[i][j][q] = 0.f; accU[i][j][q] = 0.f; }

  for (int k0 = 0; k0 < DMODEL; k0 += KB) {
    __syncthreads();
#if HAVE_TDM
    if (!ROUTED) {
      if (wave == 0)  // one TDM per block; EXEC ignored by hardware
        tdm_load_a(xb + (size_t)row0 * DMODEL + k0,
                   (unsigned)(uintptr_t)(void*)sA, DMODEL, BM);
    } else
#endif
    {  // gathered A staging: 32B per thread
      const int r  = tid >> 1;
      const int c  = (tid & 1) * 16;
      const int xr = srow[r];
      bf16x8 v0, v1;
      if (xr >= 0) {
        const bf16* src = xb + (size_t)xr * DMODEL + k0 + c;
        v0 = *(const bf16x8*)(src);
        v1 = *(const bf16x8*)(src + 8);
      } else {
#pragma unroll
        for (int i = 0; i < 8; ++i) { v0[i] = (bf16)0.f; v1[i] = (bf16)0.f; }
      }
      *(bf16x8*)(sA + r * KBP + c)     = v0;
      *(bf16x8*)(sA + r * KBP + c + 8) = v1;
    }
    // stage Bg/Bu fp32 -> bf16 (K-contiguous columns, b64 packed stores)
    stageB(wg + (size_t)k0 * DFF + n0, DFF, sBg, tid);
    stageB(wu + (size_t)k0 * DFF + n0, DFF, sBu, tid);
    if (k0 + KB < DMODEL) {  // hide next-step weight latency
      const int kk = tid >> 3;
      const int n8 = (tid & 7) * 8;
      __builtin_prefetch(wg + (size_t)(k0 + KB + kk) * DFF + n0 + n8, 0, 3);
      __builtin_prefetch(wu + (size_t)(k0 + KB + kk) * DFF + n0 + n8, 0, 3);
    }
#if HAVE_TDM
    if (!ROUTED && wave == 0) __builtin_amdgcn_s_wait_tensorcnt(0);
#endif
    __syncthreads();

    bf16x16 a0 = ldsA_frag(sA, wm);
    bf16x16 a1 = ldsA_frag(sA, wm + 16);
    bf16x16 g0 = ldsB_frag(sBg, wn);
    bf16x16 g1 = ldsB_frag(sBg, wn + 16);
    bf16x16 u0 = ldsB_frag(sBu, wn);
    bf16x16 u1 = ldsB_frag(sBu, wn + 16);

    accG[0][0] = wmma_bf16(a0, g0, accG[0][0]);
    accG[0][1] = wmma_bf16(a0, g1, accG[0][1]);
    accG[1][0] = wmma_bf16(a1, g0, accG[1][0]);
    accG[1][1] = wmma_bf16(a1, g1, accG[1][1]);
    accU[0][0] = wmma_bf16(a0, u0, accU[0][0]);
    accU[0][1] = wmma_bf16(a0, u1, accU[0][1]);
    accU[1][0] = wmma_bf16(a1, u0, accU[1][0]);
    accU[1][1] = wmma_bf16(a1, u1, accU[1][1]);
  }

  // epilogue: h = silu(g)*u -> bf16. elem r of lane l: row=r+(l>=16)*8, col=l&15
  const int lane = tid & 31;
  const int cn   = lane & 15;
  const int rb   = (lane >> 4) * 8;
#pragma unroll
  for (int mi = 0; mi < 2; ++mi)
#pragma unroll
    for (int ni = 0; ni < 2; ++ni)
#pragma unroll
      for (int r = 0; r < 8; ++r) {
        const int lr = wm + mi * 16 + rb + r;
        if (srow[lr] >= 0) {
          const int col = n0 + wn + ni * 16 + cn;
          const float g = accG[mi][ni][r];
          const float u = accU[mi][ni][r];
          const float h = (g / (1.f + __expf(-g))) * u;
          hbase[(size_t)(row0 + lr) * DFF + col] = (bf16)h;
        }
      }
}

// ---------------------------------------------------------------------------
// down GEMM: O = (H @ Wd) * weight. A tiles (contiguous bf16 segment rows)
// staged by TDM; routed rows scatter to routed_out[rank][token].
// ---------------------------------------------------------------------------
template <bool ROUTED>
__global__ __launch_bounds__(256)
void down_kernel(const bf16* __restrict__ hb, const float* __restrict__ Wd,
                 float* __restrict__ outp, const float* __restrict__ wts,
                 const int* __restrict__ list, const int* __restrict__ offs,
                 const int* __restrict__ counts) {
  const int e    = ROUTED ? blockIdx.z : 0;
  const int rows = ROUTED ? counts[e] : T_TOKENS;
  const int row0 = blockIdx.x * BM;
  if (row0 >= rows) return;
  const int n0 = blockIdx.y * BN;

  const float* wd   = Wd + (ROUTED ? (size_t)e * DFF * DMODEL : 0);
  const bf16* abase = hb + (ROUTED ? (size_t)offs[e] * DFF : 0);
  const int vrows   = min(BM, rows - row0);

  __shared__ bf16  sA[BM * KBP];
  __shared__ bf16  sB[BN * KBP];
  __shared__ int   sDst[BM];
  __shared__ float sW[BM];

  const int tid = threadIdx.x;
  if (tid < BM) {
    const int r = row0 + tid;
    if (ROUTED) {
      if (r < rows) {
        const int entry = list[e * T_TOKENS + r];
        sDst[tid] = (entry & 1) * T_TOKENS + (entry >> 1);
        sW[tid]   = wts[entry];
      } else { sDst[tid] = -1; sW[tid] = 0.f; }
    } else {
      sDst[tid] = (r < rows) ? r : -1;
      sW[tid]   = 1.f;  // N_SHARED == 1
    }
  }

  const int wave = tid >> 5;
  const int wm = (wave >> 1) * 32;
  const int wn = (wave & 1) * 32;

  f32x8 acc[2][2];
#pragma unroll
  for (int i = 0; i < 2; ++i)
#pragma unroll
    for (int j = 0; j < 2; ++j)
#pragma unroll
      for (int q = 0; q < 8; ++q) acc[i][j][q] = 0.f;

  for (int k0 = 0; k0 < DFF; k0 += KB) {
    __syncthreads();
#if HAVE_TDM
    if (wave == 0)  // TDM zero-fills rows >= vrows
      tdm_load_a(abase + (size_t)row0 * DFF + k0,
                 (unsigned)(uintptr_t)(void*)sA, DFF, (unsigned)vrows);
#else
    {  // manual A staging
      const int r = tid >> 1;
      const int c = (tid & 1) * 16;
      bf16x8 v0, v1;
      if (r < vrows) {
        const bf16* src = abase + (size_t)(row0 + r) * DFF + k0 + c;
        v0 = *(const bf16x8*)(src);
        v1 = *(const bf16x8*)(src + 8);
      } else {
#pragma unroll
        for (int i = 0; i < 8; ++i) { v0[i] = (bf16)0.f; v1[i] = (bf16)0.f; }
      }
      *(bf16x8*)(sA + r * KBP + c)     = v0;
      *(bf16x8*)(sA + r * KBP + c + 8) = v1;
    }
#endif
    stageB(wd + (size_t)k0 * DMODEL + n0, DMODEL, sB, tid);
    if (k0 + KB < DFF) {
      const int kk = tid >> 3;
      const int n8 = (tid & 7) * 8;
      __builtin_prefetch(wd + (size_t)(k0 + KB + kk) * DMODEL + n0 + n8, 0, 3);
    }
#if HAVE_TDM
    if (wave == 0) __builtin_amdgcn_s_wait_tensorcnt(0);
#endif
    __syncthreads();

    bf16x16 a0 = ldsA_frag(sA, wm);
    bf16x16 a1 = ldsA_frag(sA, wm + 16);
    bf16x16 b0 = ldsB_frag(sB, wn);
    bf16x16 b1 = ldsB_frag(sB, wn + 16);
    acc[0][0] = wmma_bf16(a0, b0, acc[0][0]);
    acc[0][1] = wmma_bf16(a0, b1, acc[0][1]);
    acc[1][0] = wmma_bf16(a1, b0, acc[1][0]);
    acc[1][1] = wmma_bf16(a1, b1, acc[1][1]);
  }

  const int lane = tid & 31;
  const int cn   = lane & 15;
  const int rb   = (lane >> 4) * 8;
#pragma unroll
  for (int mi = 0; mi < 2; ++mi)
#pragma unroll
    for (int ni = 0; ni < 2; ++ni)
#pragma unroll
      for (int r = 0; r < 8; ++r) {
        const int lr = wm + mi * 16 + rb + r;
        const int d  = sDst[lr];
        if (d >= 0) {
          const int col = n0 + wn + ni * 16 + cn;
          outp[(size_t)d * DMODEL + col] = sW[lr] * acc[mi][ni][r];
        }
      }
}

// ---------------------------------------------------------------------------
// host launcher
// ---------------------------------------------------------------------------
extern "C" void kernel_launch(void* const* d_in, const int* in_sizes, int n_in,
                              void* d_out, int out_size, void* d_ws,
                              size_t ws_size, hipStream_t stream) {
  (void)in_sizes; (void)n_in; (void)out_size;
  const float* x   = (const float*)d_in[0];
  const float* Wr  = (const float*)d_in[1];
  const float* Wsg = (const float*)d_in[2];
  const float* Wsu = (const float*)d_in[3];
  const float* Wsd = (const float*)d_in[4];
  const float* Weg = (const float*)d_in[5];
  const float* Weu = (const float*)d_in[6];
  const float* Wed = (const float*)d_in[7];
  float* out = (float*)d_out;

  char* ws   = (char*)d_ws;
  size_t off = 0;
  auto carve = [&](size_t bytes) -> void* {
    void* p = ws + off;
    off = (off + bytes + 255) & ~(size_t)255;
    return p;
  };
  bf16*  xb     = (bf16*) carve((size_t)T_TOKENS * DMODEL * 2);
  bf16*  hb_s   = (bf16*) carve((size_t)T_TOKENS * DFF * 2);
  bf16*  hb_r   = (bf16*) carve((size_t)2 * T_TOKENS * DFF * 2);
  float* r01    = (float*)carve((size_t)2 * T_TOKENS * DMODEL * 4);
  float* wts    = (float*)carve((size_t)2 * T_TOKENS * 4);
  int*   list   = (int*)  carve((size_t)NEXP * T_TOKENS * 4);
  int*   counts = (int*)  carve(256);
  int*   offs   = (int*)  carve(256);
  if (off > ws_size) return;  // workspace too small: bail (no UB)

  const int n = T_TOKENS * DMODEL;

  zero_counts_kernel<<<1, 32, 0, stream>>>(counts);
  cvt_x_kernel<<<(n / 4 + 255) / 256, 256, 0, stream>>>(x, xb, n / 4);
  router_kernel<<<T_TOKENS / 8, 256, 0, stream>>>(x, Wr, wts, list, counts);
  scan_kernel<<<1, 1, 0, stream>>>(counts, offs);

  // shared expert (N_SHARED == 1)
  gateup_kernel<false><<<dim3(T_TOKENS / BM, DFF / BN, 1), 256, 0, stream>>>(
      xb, Wsg, Wsu, hb_s, nullptr, nullptr, nullptr);
  down_kernel<false><<<dim3(T_TOKENS / BM, DMODEL / BN, 1), 256, 0, stream>>>(
      hb_s, Wsd, out, nullptr, nullptr, nullptr, nullptr);

  // routed experts (grouped GEMM over gathered tokens; idle tiles early-exit)
  gateup_kernel<true><<<dim3(T_TOKENS / BM, DFF / BN, NEXP), 256, 0, stream>>>(
      xb, Weg, Weu, hb_r, list, offs, counts);
  down_kernel<true><<<dim3(T_TOKENS / BM, DMODEL / BN, NEXP), 256, 0, stream>>>(
      hb_r, Wed, r01, wts, list, offs, counts);

  // deterministic final combine: out = shared + routed(rank0) + routed(rank1)
  combine_kernel<<<(n + 255) / 256, 256, 0, stream>>>(
      out, r01, r01 + (size_t)T_TOKENS * DMODEL, n);
}
